// TriangleAttentionStartingNode_61400852464438
// MI455X (gfx1250) — compile-verified, matrix operand order checked by reading, not run on hardware
//
#include <hip/hip_runtime.h>
#include <hip/hip_bf16.h>
#include <math.h>

// ---------------------------------------------------------------------------
// TriangleAttentionStartingNode for MI455X (gfx1250, wave32, WMMA 16x16x32 f16)
//   B=1, N=384, C=128, H=4, CH=32
// All matmuls on v_wmma_f32_16x16x32_f16; all fragment traffic as b128;
// shared K/V panels staged once per block in LDS (319488 B of 320 KB/WGP).
// ---------------------------------------------------------------------------

typedef _Float16 f16;
typedef __attribute__((ext_vector_type(16))) _Float16 v16h;
typedef __attribute__((ext_vector_type(8)))  _Float16 h8;
typedef __attribute__((ext_vector_type(4)))  _Float16 h4;
typedef __attribute__((ext_vector_type(8)))  float    v8f;

constexpr int kN   = 384;
constexpr int kC   = 128;
constexpr int kH   = 4;
constexpr int kCH  = 32;
constexpr int kNN  = kN * kN;          // 147456 positions
constexpr float kLnEps = 1e-5f;

// ---- workspace layout (bytes) ---------------------------------------------
constexpr size_t SZ_H   = (size_t)kNN * kC * sizeof(f16);   // 37,748,736
constexpr size_t OFF_XN = 0;                 // f16 [NN,128] normalized x (row major)
constexpr size_t OFF_Q  = 1 * SZ_H;          // f16 [NN,128] q (pre-scaled)
constexpr size_t OFF_K  = 2 * SZ_H;          // f16 [NN,128]
constexpr size_t OFF_VT = 3 * SZ_H;          // f16 [(i*128+e), kk] V transposed
constexpr size_t OFF_G  = 4 * SZ_H;          // f16 [NN,128] sigmoid gate
constexpr size_t OFF_O  = 5 * SZ_H;          // f16 [NN,128] attention out
constexpr size_t OFF_PB = 6 * SZ_H;          // f32 [H,N,N]  pair bias
constexpr size_t SZ_PB  = (size_t)kH * kNN * sizeof(float); // 2,359,296
constexpr size_t OFF_WT = OFF_PB + SZ_PB;    // f16 5x fragment-native weights

// ---- WMMA fragment helpers (ISA 7.12.2 16-bit layouts, wave32) -------------
// Per-lane VGPR-pair p maps to K offsets: lanes 0-15 -> {0..7,16..23},
// lanes 16-31 -> {8..15,24..31}.  Both runs are 8 contiguous halves (16 B).
__device__ __forceinline__ int frag_k_off(int p, int hl) {
  int base = (p < 4) ? (2 * p) : (16 + 2 * (p - 4));
  return base + hl * 8;
}

__device__ __forceinline__ v16h concat_h8(h8 lo, h8 hi) {
  return __builtin_shufflevector(lo, hi, 0, 1, 2, 3, 4, 5, 6, 7,
                                 8, 9, 10, 11, 12, 13, 14, 15);
}

// A fragment: rows rm..rm+15 x K kc..kc+31, row-major src, stride ld halves.
template <typename PTR>
__device__ __forceinline__ v16h load_a_frag(PTR src, int rm, int kc, int ld) {
  int lane = threadIdx.x & 31;
  int hl = lane >> 4, lr = lane & 15;
  auto row = src + (size_t)(rm + lr) * ld + kc + hl * 8;
  h8 lo = *(const h8*)(row);
  h8 hi = *(const h8*)(row + 16);
  return concat_h8(lo, hi);
}

// B fragment from a *transposed* operand stored row-major as [Ncol][K]:
// lane owns one column row (n0+lr) and reads two contiguous 16-B K runs.
template <typename PTR>
__device__ __forceinline__ v16h load_bT_frag(PTR srcT, int kc, int n0, int ld) {
  int lane = threadIdx.x & 31;
  int hl = lane >> 4, lr = lane & 15;
  auto row = srcT + (size_t)(n0 + lr) * ld + kc + hl * 8;
  h8 lo = *(const h8*)(row);
  h8 hi = *(const h8*)(row + 16);
  return concat_h8(lo, hi);
}

// B fragment from fragment-native weight storage: one contiguous v16h/lane.
__device__ __forceinline__ v16h load_bW_frag(const f16* wt, int m, int nt, int kc) {
  int lane = threadIdx.x & 31;
  const v16h* p = (const v16h*)wt + ((((size_t)m * 8 + nt) * 4 + kc) * 32 + lane);
  return *p;
}

__device__ __forceinline__ v8f wmma_f16(v16h a, v16h b, v8f c) {
  return __builtin_amdgcn_wmma_f32_16x16x32_f16(false, a, false, b,
                                                (short)0, c, false, false);
}

// ---------------------------------------------------------------------------
// Kernel 0: pack the five 128x128 f32 weights into fragment-native f16.
// ---------------------------------------------------------------------------
__global__ void wprep_kernel(const float* __restrict__ wq,
                             const float* __restrict__ wk,
                             const float* __restrict__ wv,
                             const float* __restrict__ wg,
                             const float* __restrict__ wo,
                             f16* __restrict__ wt) {
  int idx = blockIdx.x * 256 + threadIdx.x;   // < 5*16384 halves
  int m = idx >> 14;
  int t = idx & 16383;
  int el = t & 15;
  int L  = (t >> 4) & 31;
  int kc = (t >> 9) & 3;
  int nt = t >> 11;
  int p  = el >> 1, lo = el & 1;
  int hl = L >> 4,  lr = L & 15;
  int k = kc * 32 + frag_k_off(p, hl) + lo;   // contraction index (C)
  int n = nt * 16 + lr;                        // output channel
  const float* W = (m == 0) ? wq : (m == 1) ? wk : (m == 2) ? wv
                 : (m == 3) ? wg : wo;
  wt[idx] = (f16)W[n * 128 + k];
}

// ---------------------------------------------------------------------------
// Kernel 1: LayerNorm over C=128 (one wave per position) + fused pair bias
// ---------------------------------------------------------------------------
__global__ void ln_kernel(const float* __restrict__ act,
                          const float* __restrict__ ln_w,
                          const float* __restrict__ ln_b,
                          const float* __restrict__ w2d,
                          f16* __restrict__ xnh,
                          float* __restrict__ pb) {
  int tid = threadIdx.x, wave = tid >> 5, lane = tid & 31;
  size_t p = (size_t)blockIdx.x * 8 + wave;        // position (j*N + k)
  int cb = lane * 4;
  float4 x = *(const float4*)(act + p * kC + cb);

  float s = x.x + x.y + x.z + x.w;
#pragma unroll
  for (int m = 16; m >= 1; m >>= 1) s += __shfl_xor(s, m, 32);
  float mu = s * (1.0f / kC);
  float d0 = x.x - mu, d1 = x.y - mu, d2 = x.z - mu, d3 = x.w - mu;
  float v = d0 * d0 + d1 * d1 + d2 * d2 + d3 * d3;
#pragma unroll
  for (int m = 16; m >= 1; m >>= 1) v += __shfl_xor(v, m, 32);
  float rstd = rsqrtf(v * (1.0f / kC) + kLnEps);

  float4 w = *(const float4*)(ln_w + cb);
  float4 b = *(const float4*)(ln_b + cb);
  float y0 = d0 * rstd * w.x + b.x;
  float y1 = d1 * rstd * w.y + b.y;
  float y2 = d2 * rstd * w.z + b.z;
  float y3 = d3 * rstd * w.w + b.w;

  h4 yh; yh[0] = (f16)y0; yh[1] = (f16)y1; yh[2] = (f16)y2; yh[3] = (f16)y3;
  *(h4*)(xnh + p * kC + cb) = yh;

#pragma unroll
  for (int hh = 0; hh < kH; ++hh) {
    float4 ww = *(const float4*)(w2d + hh * kC + cb);
    float d = y0 * ww.x + y1 * ww.y + y2 * ww.z + y3 * ww.w;
#pragma unroll
    for (int m = 16; m >= 1; m >>= 1) d += __shfl_xor(d, m, 32);
    if (lane == 0) pb[(size_t)hh * kNN + p] = d;
  }
}

// ---------------------------------------------------------------------------
// Kernel 2: Q/K/V/G projections, templated on MODE (no dynamic branching).
// MODE 0: q (scaled), 1: k, 3: gate (sigmoid via v_rcp) -> LDS-staged stores.
// MODE 2: v -> stored TRANSPOSED (vt[(i*128+e)*384 + kk]) straight from the
//         D fragment: 8 consecutive kk per lane = one 16-B store.
// ---------------------------------------------------------------------------
template <int MODE>
__global__ void qkvg_kernel(const f16* __restrict__ xnh,
                            const f16* __restrict__ wt,
                            const float* __restrict__ bg,
                            f16* __restrict__ out) {
  __shared__ f16 stile[128 * 128];   // 32 KB staging tile (unused for MODE 2)

  int tid = threadIdx.x, wave = tid >> 5, lane = tid & 31;
  int hl = lane >> 4, lr = lane & 15;
  int rm = blockIdx.x * 128 + wave * 16;
  int i_row = rm / kN;               // hoisted; tiles never cross an i row
  int kk0   = rm - i_row * kN + hl * 8;

  v16h a[4];
#pragma unroll
  for (int kc = 0; kc < 4; ++kc) a[kc] = load_a_frag(xnh, rm, kc * 32, kC);

  const float qscale = 0.17677669529663687f;  // 1/sqrt(32)
#pragma unroll
  for (int nt = 0; nt < 8; ++nt) {
    int n0 = nt * 16;
    v8f acc = {};
#pragma unroll
    for (int kc = 0; kc < 4; ++kc)
      acc = wmma_f16(a[kc], load_bW_frag(wt, MODE, nt, kc), acc);

    int col = n0 + lr;
    if constexpr (MODE == 2) {
      h8 pk;
#pragma unroll
      for (int r = 0; r < 8; ++r) pk[r] = (f16)acc[r];
      *(h8*)(out + ((size_t)i_row * 128 + col) * kN + kk0) = pk;
    } else {
#pragma unroll
      for (int r = 0; r < 8; ++r) {
        float v = acc[r];
        if constexpr (MODE == 0) v *= qscale;
        if constexpr (MODE == 3)
          v = __builtin_amdgcn_rcpf(1.0f + __expf(-(v + bg[col])));
        stile[(wave * 16 + hl * 8 + r) * 128 + col] = (f16)v;
      }
    }
  }

  if constexpr (MODE != 2) {
    __syncthreads();
    int row = tid >> 1, hv = tid & 1;     // 128 B per thread, coalesced
    const h8* src = (const h8*)(stile + row * 128 + hv * 64);
    h8* dst = (h8*)(out + ((size_t)blockIdx.x * 128 + row) * kC + hv * 64);
#pragma unroll
    for (int q = 0; q < 8; ++q) dst[q] = src[q];
  }
}

// ---------------------------------------------------------------------------
// Kernel 3: per-row multi-head attention.
// grid=(3 j-blocks, 4 heads, 384 i); block=256 (8 waves).
// Dynamic LDS (319488 B of 320 KB/WGP):
//   S  f32[128][384]  196608 B   logits / exp / O-staging scratch
//   P  f16[128][384]   98304 B   softmax probabilities
//   KV f16 24576 B              K panel (pass 1) then V^T panel (pass 3)
// Staging the shared K / V^T panels once per block cuts their global traffic
// 8x (each of the 8 waves previously re-fetched the same 24 KB panel).
// ---------------------------------------------------------------------------
__global__ void attn_kernel(const f16* __restrict__ qh,
                            const f16* __restrict__ kh,
                            const f16* __restrict__ vt,
                            const float* __restrict__ pb,
                            const float* __restrict__ mask,
                            f16* __restrict__ oh) {
  extern __shared__ char smem[];
  float* S  = (float*)smem;                                   // [128][384]
  f16*   P  = (f16*)(smem + 128 * kN * sizeof(float));        // [128][384]
  f16*   KV = (f16*)(smem + 128 * kN * (sizeof(float) + sizeof(f16)));

  int jb = blockIdx.x, hd = blockIdx.y, i = blockIdx.z;
  int tid = threadIdx.x, wave = tid >> 5, lane = tid & 31;
  int hl = lane >> 4, lr = lane & 15;
  int j0 = jb * 128;
  int jr = j0 + wave * 16;
  const size_t rowbase = (size_t)i * kN * kC + (size_t)hd * kCH;
  const f16* vtb = vt + ((size_t)i * 128 + (size_t)hd * kCH) * kN;

  // warm L2 with the V^T panel (speculative global_prefetch_b8)
  __builtin_prefetch(vtb + (size_t)tid * 48, 0, 1);

  // ---- stage K panel [384 kk][32 d] into LDS (b128 both sides) ------------
  for (int c = tid; c < kN * 4; c += 256) {     // 1536 16-B chunks
    int kk = c >> 2, off = (c & 3) * 8;
    *(h8*)(KV + kk * kCH + off) =
        *(const h8*)(kh + rowbase + (size_t)kk * kC + off);
  }
  __syncthreads();

  // Q fragment persists across k tiles (CH == 32 == one WMMA K step)
  v16h qa = load_a_frag(qh + rowbase, jr, 0, kC);

  // ---- pass 1: raw logits into LDS ----------------------------------------
  for (int kt = 0; kt < kN / 16; ++kt) {
    int k0 = kt * 16;
    v16h kb = load_bT_frag(KV, 0, k0, kCH);     // 2x ds_load_b128
    v8f c = {};
    c = wmma_f16(qa, kb, c);
    int col = k0 + lr;
#pragma unroll
    for (int r = 0; r < 8; ++r) {
      int jl = wave * 16 + hl * 8 + r;
      S[jl * kN + col] = c[r];
    }
  }
  __syncthreads();

  // ---- pass 2: biases + softmax (2 threads per row, float4 sweeps) --------
  {
    int row = tid >> 1;
    int hv  = tid & 1;
    int jrow = j0 + row;
    float* srow = S + row * kN;
    const float* pbrow = pb + (size_t)hd * kNN + (size_t)jrow * kN;
    const float* mrow  = mask + (size_t)i * kN;
    int x0 = hv * 192;

    float m = -1e30f;
    for (int x = x0; x < x0 + 192; x += 4) {
      float4 s  = *(float4*)(srow + x);
      float4 pv = *(const float4*)(pbrow + x);
      float4 mv = *(const float4*)(mrow + x);
      s.x += pv.x + 1e9f * (mv.x - 1.0f);
      s.y += pv.y + 1e9f * (mv.y - 1.0f);
      s.z += pv.z + 1e9f * (mv.z - 1.0f);
      s.w += pv.w + 1e9f * (mv.w - 1.0f);
      *(float4*)(srow + x) = s;
      m = fmaxf(m, fmaxf(fmaxf(s.x, s.y), fmaxf(s.z, s.w)));
    }
    m = fmaxf(m, __shfl_xor(m, 1, 32));

    float sum = 0.f;
    for (int x = x0; x < x0 + 192; x += 4) {
      float4 s = *(float4*)(srow + x);
      s.x = __expf(s.x - m); s.y = __expf(s.y - m);
      s.z = __expf(s.z - m); s.w = __expf(s.w - m);
      *(float4*)(srow + x) = s;
      sum += s.x + s.y + s.z + s.w;
    }
    sum += __shfl_xor(sum, 1, 32);
    float inv = __builtin_amdgcn_rcpf(sum);

    f16* prow = P + row * kN;
    for (int x = x0; x < x0 + 192; x += 4) {
      float4 s = *(float4*)(srow + x);
      h4 ph;
      ph[0] = (f16)(s.x * inv); ph[1] = (f16)(s.y * inv);
      ph[2] = (f16)(s.z * inv); ph[3] = (f16)(s.w * inv);
      *(h4*)(prow + x) = ph;
    }
  }

  // ---- stage V^T panel [32 d][384 kk] into the (now free) K slot ----------
  for (int c = tid; c < kCH * 48; c += 256) {   // 1536 16-B chunks, coalesced
    int d = c / 48, off = (c % 48) * 8;
    *(h8*)(KV + d * kN + off) = *(const h8*)(vtb + (size_t)d * kN + off);
  }
  __syncthreads();

  // ---- pass 3: O = P x V^T  (16 rows x 32 d per wave, 12 K=32 steps) ------
  v8f acc0 = {}, acc1 = {};
  for (int kc = 0; kc < kN; kc += 32) {
    v16h pa  = load_a_frag(P, wave * 16, kc, kN);  // 2x ds_load_b128
    v16h vb0 = load_bT_frag(KV, kc, 0,  kN);       // 2x ds_load_b128
    v16h vb1 = load_bT_frag(KV, kc, 16, kN);
    acc0 = wmma_f16(pa, vb0, acc0);
    acc1 = wmma_f16(pa, vb1, acc1);
  }

  // ---- epilogue: stage in LDS (reuse S region), coalesced global writes ---
  f16* OT = (f16*)S;                                  // [128][32]
#pragma unroll
  for (int r = 0; r < 8; ++r) {
    int jl = wave * 16 + hl * 8 + r;
    OT[jl * kCH + lr]      = (f16)acc0[r];
    OT[jl * kCH + 16 + lr] = (f16)acc1[r];
  }
  __syncthreads();
  {
    int j = tid >> 1, hv = tid & 1;                  // 32 B per thread
    const h8* src = (const h8*)(OT + j * kCH + hv * 16);
    h8* dst = (h8*)(oh + ((size_t)i * kN + j0 + j) * kC + hd * kCH + hv * 16);
    dst[0] = src[0];
    dst[1] = src[1];
  }
}

// ---------------------------------------------------------------------------
// Kernel 4: out = (o * g) @ wo^T + bo   (f32 output)
// ---------------------------------------------------------------------------
__global__ void out_kernel(const f16* __restrict__ oh,
                           const f16* __restrict__ gh,
                           const f16* __restrict__ wt,    // fragment-native, m=4
                           const float* __restrict__ bo,
                           float* __restrict__ out) {
  int tid = threadIdx.x, wave = tid >> 5, lane = tid & 31;
  int hl = lane >> 4, lr = lane & 15;
  int rm = blockIdx.x * 128 + wave * 16;

  v16h a[4];
#pragma unroll
  for (int kc = 0; kc < 4; ++kc) {
    const f16* orow = oh + ((size_t)rm + lr) * kC + kc * 32 + hl * 8;
    const f16* grow = gh + ((size_t)rm + lr) * kC + kc * 32 + hl * 8;
    h8 olo = *(const h8*)orow,        ohi = *(const h8*)(orow + 16);
    h8 glo = *(const h8*)grow,        ghi = *(const h8*)(grow + 16);
    a[kc] = concat_h8(olo * glo, ohi * ghi);   // packed f16 multiply
  }

#pragma unroll
  for (int nt = 0; nt < 8; ++nt) {
    int n0 = nt * 16;
    v8f acc = {};
#pragma unroll
    for (int kc = 0; kc < 4; ++kc)
      acc = wmma_f16(a[kc], load_bW_frag(wt, 4, nt, kc), acc);
    int col = n0 + lr;
    float bias = bo[col];
#pragma unroll
    for (int r = 0; r < 8; ++r) {
      size_t p = (size_t)rm + hl * 8 + r;
      out[p * kC + col] = acc[r] + bias;
    }
  }
}

// ---------------------------------------------------------------------------
extern "C" void kernel_launch(void* const* d_in, const int* in_sizes, int n_in,
                              void* d_out, int out_size, void* d_ws, size_t ws_size,
                              hipStream_t stream) {
  const float* act  = (const float*)d_in[0];
  const float* mask = (const float*)d_in[1];
  const float* ln_w = (const float*)d_in[2];
  const float* ln_b = (const float*)d_in[3];
  const float* w2d  = (const float*)d_in[4];
  const float* wq   = (const float*)d_in[5];
  const float* wk   = (const float*)d_in[6];
  const float* wv   = (const float*)d_in[7];
  const float* wg   = (const float*)d_in[8];
  const float* bg   = (const float*)d_in[9];
  const float* wo   = (const float*)d_in[10];
  const float* bo   = (const float*)d_in[11];
  float* out = (float*)d_out;

  char* ws = (char*)d_ws;
  f16*   xnh = (f16*)(ws + OFF_XN);
  f16*   qh  = (f16*)(ws + OFF_Q);
  f16*   kh  = (f16*)(ws + OFF_K);
  f16*   vtb = (f16*)(ws + OFF_VT);
  f16*   gh  = (f16*)(ws + OFF_G);
  f16*   ohb = (f16*)(ws + OFF_O);
  float* pb  = (float*)(ws + OFF_PB);
  f16*   wt  = (f16*)(ws + OFF_WT);

  wprep_kernel<<<(5 * 16384) / 256, 256, 0, stream>>>(wq, wk, wv, wg, wo, wt);

  ln_kernel<<<kNN / 8, 256, 0, stream>>>(act, ln_w, ln_b, w2d, xnh, pb);

  qkvg_kernel<0><<<kNN / 128, 256, 0, stream>>>(xnh, wt, bg, qh);
  qkvg_kernel<1><<<kNN / 128, 256, 0, stream>>>(xnh, wt, bg, kh);
  qkvg_kernel<2><<<kNN / 128, 256, 0, stream>>>(xnh, wt, bg, vtb);
  qkvg_kernel<3><<<kNN / 128, 256, 0, stream>>>(xnh, wt, bg, gh);

  constexpr size_t kAttnLds =
      128 * kN * (sizeof(float) + sizeof(f16)) + (size_t)kN * kCH * sizeof(f16);
  (void)hipFuncSetAttribute((const void*)attn_kernel,
                            hipFuncAttributeMaxDynamicSharedMemorySize,
                            (int)kAttnLds);
  attn_kernel<<<dim3(kN / 128, kH, kN), 256, kAttnLds, stream>>>(
      qh, kh, vtb, pb, mask, ohb);

  out_kernel<<<kNN / 128, 256, 0, stream>>>(ohb, gh, wt, bo, out);
}
